// RegionGrowingModule_8040178778374
// MI455X (gfx1250) — compile-verified, matrix operand order checked by reading, not run on hardware
//
#include <hip/hip_runtime.h>

#define HH 768
#define WW 768
#define NPIX (HH*WW)
#define NIMG 8
#define THRESH_V 0.1563f   // 0.0789 + 1.0*0.0774

typedef _Float16 v16h __attribute__((ext_vector_type(16)));
typedef _Float16 h8   __attribute__((ext_vector_type(8)));
typedef float    v8f  __attribute__((ext_vector_type(8)));

__device__ __forceinline__ int reflect101(int i, int n){ if(i<0) i=-i; if(i>=n) i=2*n-2-i; return i; }
__device__ __forceinline__ int symref(int i, int n){ if(i<0) i=-i-1; if(i>=n) i=2*n-1-i; return i; }
__device__ __forceinline__ int clampi(int v, int n){ return v<0?0:(v>=n?n-1:v); }

// ---------------- filter tap generation (device, deterministic) ----------------
__global__ void init_filters_kernel(float* __restrict__ gauss, float* __restrict__ gtap){
  int t = threadIdx.x;
  if (t == 0){
    const float sig = 15.5f;                 // cv2 sigma for ksize=101, sigma=0
    float tmp[101]; float s = 0.f;
    for (int i=0;i<101;i++){ float xx=(float)i-50.f; tmp[i]=expf(-xx*xx/(2.f*sig*sig)); s+=tmp[i]; }
    for (int i=0;i<101;i++) gauss[i]=tmp[i]/s;
  }
  const float PI = 3.14159265358979f;
  float sigma = (1.0f/PI)*sqrtf(logf(2.0f)*0.5f)*3.0f/0.1f;   // ~5.6216
  float norm  = 1.f/(2.f*PI*sigma*sigma);
  for (int ff=0; ff<4; ff++){
    float ang = (45.f + 45.f*(float)ff) * PI/180.f;
    float ct = cosf(ang), st = sinf(ang);
    int x0 = (int)ceilf(fmaxf(fmaxf(fabsf(3.f*sigma*ct), fabsf(3.f*sigma*st)), 1.f));
    int S  = 2*x0+1;                          // 25 or 35
    for (int idx=t; idx<S*S; idx+=blockDim.x){
      int r=idx/S, c=idx-r*S;
      float y=(float)(r-x0), x=(float)(c-x0);
      float rx =  x*ct + y*st;
      float ry = -x*st + y*ct;
      // 180-degree flip is identity (even-symmetric real Gabor)
      gtap[ff*1225+idx] = expf(-0.5f*(rx*rx+ry*ry)/(sigma*sigma))*norm*cosf(2.f*PI*0.1f*rx);
    }
  }
}

// ---------------- precompute Toeplitz B fragments in WMMA lane layout ----------------
// bfrag[ff][frag=r*2+q][lane][e] (f16): B[K][n] with K = 16*(e>>3) + 8*(lane>>4) + (e&7),
// n = lane&15, tap col c = 32q + K - n, zero outside [0,S).
__global__ __launch_bounds__(256) void init_bfrag_kernel(const float* __restrict__ gtap,
                                                         _Float16* __restrict__ bfrag){
  int ff = blockIdx.x;
  const int Stab[4] = {25,35,25,35};
  int S = Stab[ff];
  const float* tp = gtap + ff*1225;
  _Float16* bf = bfrag + (size_t)ff*70*512;
  for (int idx=threadIdx.x; idx < S*2*512; idx+=256){
    int frag = idx >> 9;          // r*2+q
    int rem  = idx & 511;
    int lane = rem >> 4;
    int e    = rem & 15;
    int r = frag >> 1, q = frag & 1;
    int K = ((e>>3)<<4) + ((lane>>4)<<3) + (e&7);
    int n = lane & 15;
    int c = (q<<5) + K - n;
    float v = (c>=0 && c<S) ? tp[r*S+c] : 0.f;
    bf[idx] = (_Float16)v;
  }
}

// ---------------- per-image min/max (deterministic tree reduce) ----------------
__global__ __launch_bounds__(1024) void minmax_kernel(const float* __restrict__ x,
                                                      float* __restrict__ mn_out,
                                                      float* __restrict__ mx_out){
  int im = blockIdx.x;
  const float* p = x + (size_t)im*NPIX;
  float mn = 1e30f, mx = -1e30f;
  for (int i=threadIdx.x; i<NPIX; i+=1024){ float v=p[i]; mn=fminf(mn,v); mx=fmaxf(mx,v); }
  __shared__ float smn[1024], smx[1024];
  smn[threadIdx.x]=mn; smx[threadIdx.x]=mx; __syncthreads();
  for (int s=512; s>0; s>>=1){
    if (threadIdx.x<s){
      smn[threadIdx.x]=fminf(smn[threadIdx.x],smn[threadIdx.x+s]);
      smx[threadIdx.x]=fmaxf(smx[threadIdx.x],smx[threadIdx.x+s]);
    }
    __syncthreads();
  }
  if (threadIdx.x==0){ mn_out[im]=smn[0]; mx_out[im]=smx[0]; }
}

__global__ __launch_bounds__(1024) void mean_kernel(const float* __restrict__ x,
                                                    float* __restrict__ mean_out){
  int im = blockIdx.x;
  const float* p = x + (size_t)im*NPIX;
  float s = 0.f;
  for (int i=threadIdx.x; i<NPIX; i+=1024) s += p[i];
  __shared__ float sh[1024];
  sh[threadIdx.x]=s; __syncthreads();
  for (int st=512; st>0; st>>=1){ if (threadIdx.x<st) sh[threadIdx.x]+=sh[threadIdx.x+st]; __syncthreads(); }
  if (threadIdx.x==0) mean_out[im] = sh[0]*(1.0f/(float)NPIX);
}

// ---------------- separable 101-tap Gaussian, reflect-101 ----------------
__global__ __launch_bounds__(256) void gauss_h_kernel(const float* __restrict__ img,
                                                      const float* __restrict__ scal,
                                                      const float* __restrict__ g,
                                                      float* __restrict__ out){
  __shared__ float s[356];
  __shared__ float gk[101];
  int im = blockIdx.z, y = blockIdx.y, xb = blockIdx.x*256;
  float mn = scal[im], inv = 1.f/(scal[8+im]-mn);
  const float* row = img + (size_t)im*NPIX + (size_t)y*WW;
  for (int i=threadIdx.x; i<356; i+=256){
    int xx = reflect101(xb + i - 50, WW);
    s[i] = (row[xx]-mn)*inv;
  }
  if (threadIdx.x < 101) gk[threadIdx.x] = g[threadIdx.x];
  __syncthreads();
  float acc = 0.f;
  #pragma unroll 4
  for (int t=0;t<101;t++) acc += gk[t]*s[threadIdx.x+t];
  out[(size_t)im*NPIX + (size_t)y*WW + xb + threadIdx.x] = acc;
}

__global__ __launch_bounds__(256) void gauss_v_kernel(const float* __restrict__ img,
                                                      const float* __restrict__ tmp,
                                                      const float* __restrict__ scal,
                                                      const float* __restrict__ g,
                                                      float* __restrict__ enh){
  int im = blockIdx.z, y = blockIdx.y;
  int x = blockIdx.x*256 + threadIdx.x;
  const float* tb = tmp + (size_t)im*NPIX;
  float acc = 0.f;
  for (int t=0;t<101;t++){
    int yy = reflect101(y + t - 50, HH);
    acc += g[t]*tb[(size_t)yy*WW + x];
  }
  float mn=scal[im], inv=1.f/(scal[8+im]-mn);
  float nv = (img[(size_t)im*NPIX + (size_t)y*WW + x]-mn)*inv;
  float e  = fminf(fmaxf(nv - acc, 0.f), 1.f);
  enh[(size_t)im*NPIX + (size_t)y*WW + x] = e;
}

// ---------------- Gabor 2-D convolution via Toeplitz WMMA ----------------
// One wave computes a 16x64 output strip (4 accumulators). Per (q,r):
//   - B fragment: one v16h per lane, straight coalesced load from precomputed
//     bfrag (same address for all blocks -> L2-resident), zero VALU.
//   - A fragments: 5 overlapping 16-B-aligned h8 LDS loads feed 4 WMMAs
//     (tile j uses runs at col 16j+32q+8hi and +16).
__global__ __launch_bounds__(32) void gabor_wmma_kernel(const float* __restrict__ enh,
                                                        const _Float16* __restrict__ bfrag,
                                                        int S,
                                                        float* __restrict__ fout){
  __shared__ __align__(16) _Float16 tile[50][112];  // (16+S-1) rows, (64+S-1) valid cols, zero pad
  const int R   = S >> 1;
  const int THt = 16 + S - 1;        // 40 or 50
  const int TWt = 64 + S - 1;        // 88 or 98
  int im = blockIdx.z;
  int y0 = blockIdx.y*16, x0 = blockIdx.x*64;
  const float* eb = enh + (size_t)im*NPIX;
  int lane = threadIdx.x;

  for (int idx=lane; idx<THt*112; idx+=32){
    int r = idx/112, c = idx - r*112;
    float v = 0.f;
    if (c < TWt){
      int gy = symref(y0 + r - R, HH);
      int gx = symref(x0 + c - R, WW);
      v = eb[(size_t)gy*WW + gx];
    }
    tile[r][c] = (_Float16)v;        // zero pad keeps NaNs out of the A operand
  }
  __syncthreads();

  int m = lane & 15, hi = lane >> 4, n = m;
  v8f z = {0.f,0.f,0.f,0.f,0.f,0.f,0.f,0.f};
  v8f acc0=z, acc1=z, acc2=z, acc3=z;

  for (int q=0; q<2; q++){
    for (int r=0; r<S; r++){
      const _Float16* bp = bfrag + ((size_t)(r*2+q)*32 + lane)*16;
      v16h b = *(const v16h*)bp;                        // 32B/lane, coalesced, L2 hit
      const _Float16* rowp = &tile[m + r][(q<<5) + (hi<<3)];
      h8 L0 = *(const h8*)(rowp);
      h8 L1 = *(const h8*)(rowp + 16);
      h8 L2 = *(const h8*)(rowp + 32);
      h8 L3 = *(const h8*)(rowp + 48);
      h8 L4 = *(const h8*)(rowp + 64);
      v16h a0 = __builtin_shufflevector(L0, L1, 0,1,2,3,4,5,6,7,8,9,10,11,12,13,14,15);
      v16h a1 = __builtin_shufflevector(L1, L2, 0,1,2,3,4,5,6,7,8,9,10,11,12,13,14,15);
      v16h a2 = __builtin_shufflevector(L2, L3, 0,1,2,3,4,5,6,7,8,9,10,11,12,13,14,15);
      v16h a3 = __builtin_shufflevector(L3, L4, 0,1,2,3,4,5,6,7,8,9,10,11,12,13,14,15);
      acc0 = __builtin_amdgcn_wmma_f32_16x16x32_f16(false, a0, false, b, (short)0, acc0, false, false);
      acc1 = __builtin_amdgcn_wmma_f32_16x16x32_f16(false, a1, false, b, (short)0, acc1, false, false);
      acc2 = __builtin_amdgcn_wmma_f32_16x16x32_f16(false, a2, false, b, (short)0, acc2, false, false);
      acc3 = __builtin_amdgcn_wmma_f32_16x16x32_f16(false, a3, false, b, (short)0, acc3, false, false);
    }
  }

  float* ob = fout + (size_t)im*NPIX;
  #pragma unroll
  for (int vv=0; vv<8; vv++){
    size_t rowo = (size_t)(y0 + vv + (hi<<3))*WW + x0 + n;
    ob[rowo     ] = acc0[vv];
    ob[rowo + 16] = acc1[vv];
    ob[rowo + 32] = acc2[vv];
    ob[rowo + 48] = acc3[vv];
  }
}

// ---------------- seeds: local 3x3 max (symmetric==clamp at radius 1) & > mean ----------------
__global__ __launch_bounds__(256) void seed_kernel(const float* __restrict__ f,
                                                   const float* __restrict__ meanv,
                                                   float* __restrict__ seed, int first){
  int im = blockIdx.y;
  int i  = blockIdx.x*256 + threadIdx.x;
  int y = i / WW, x = i - y*WW;
  const float* fb = f + (size_t)im*NPIX;
  float c = fb[i];
  float mx = -1e30f;
  #pragma unroll
  for (int dy=-1; dy<=1; dy++){
    int yy = clampi(y+dy, HH);
    #pragma unroll
    for (int dx=-1; dx<=1; dx++){
      int xx = clampi(x+dx, WW);
      mx = fmaxf(mx, fb[(size_t)yy*WW+xx]);
    }
  }
  float sv = ((c == mx) && (c > meanv[im])) ? 1.f : 0.f;
  size_t idx = (size_t)im*NPIX + i;
  seed[idx] = first ? sv : fmaxf(seed[idx], sv);
}

// ---------------- Sobel gradient magnitude -> mask ----------------
__global__ __launch_bounds__(256) void sobel_kernel(const float* __restrict__ enh,
                                                    float* __restrict__ mask){
  int im = blockIdx.y;
  int i  = blockIdx.x*256 + threadIdx.x;
  int y = i / WW, x = i - y*WW;
  const float* e = enh + (size_t)im*NPIX;
  int ym=reflect101(y-1,HH), yp=reflect101(y+1,HH);
  int xm=reflect101(x-1,WW), xp=reflect101(x+1,WW);
  float a00=e[(size_t)ym*WW+xm], a01=e[(size_t)ym*WW+x], a02=e[(size_t)ym*WW+xp];
  float a10=e[(size_t)y*WW+xm],                          a12=e[(size_t)y*WW+xp];
  float a20=e[(size_t)yp*WW+xm], a21=e[(size_t)yp*WW+x], a22=e[(size_t)yp*WW+xp];
  float gx = (a02 + 2.f*a12 + a22) - (a00 + 2.f*a10 + a20);
  float gy = (a20 + 2.f*a21 + a22) - (a00 + 2.f*a01 + a02);
  float grad = sqrtf(gx*gx + gy*gy);
  mask[(size_t)im*NPIX + i] = (grad >= THRESH_V) ? 1.f : 0.f;
}

__global__ __launch_bounds__(256) void reach0_kernel(const float* __restrict__ seed,
                                                     const float* __restrict__ mask,
                                                     float* __restrict__ reach){
  int im = blockIdx.y;
  size_t idx = (size_t)im*NPIX + blockIdx.x*256 + threadIdx.x;
  reach[idx] = seed[idx]*mask[idx];
}

// ---------------- flood fill: 64x64 tile-local Jacobi fixpoint per pass ----------------
__global__ __launch_bounds__(256) void flood_kernel(const float* __restrict__ rin,
                                                    const float* __restrict__ mask,
                                                    float* __restrict__ rout){
  __shared__ float rt[66][68];
  __shared__ float mt[64][68];
  __shared__ int changed;
  int im = blockIdx.z;
  int ty0 = blockIdx.y*64, tx0 = blockIdx.x*64;
  const float* rb = rin  + (size_t)im*NPIX;
  const float* mb = mask + (size_t)im*NPIX;
  for (int idx=threadIdx.x; idx<66*66; idx+=256){
    int r = idx/66, c = idx - r*66;
    int gy = ty0 + r - 1, gx = tx0 + c - 1;
    rt[r][c] = (gy>=0 && gy<HH && gx>=0 && gx<WW) ? rb[(size_t)gy*WW+gx] : 0.f;  // zero pad
  }
  for (int idx=threadIdx.x; idx<64*64; idx+=256){
    int r = idx>>6, c = idx&63;
    mt[r][c] = mb[(size_t)(ty0+r)*WW + tx0 + c];
  }
  __syncthreads();
  for (int it=0; it<80; it++){
    if (threadIdx.x==0) changed = 0;
    __syncthreads();
    float nv[16]; int ch = 0;
    #pragma unroll
    for (int k=0;k<16;k++){
      int p = threadIdx.x + 256*k;
      int r = p>>6, c = p&63;
      float cur = rt[r+1][c+1];
      float d = rt[r][c];
      d=fmaxf(d,rt[r][c+1]);   d=fmaxf(d,rt[r][c+2]);
      d=fmaxf(d,rt[r+1][c]);   d=fmaxf(d,rt[r+1][c+1]); d=fmaxf(d,rt[r+1][c+2]);
      d=fmaxf(d,rt[r+2][c]);   d=fmaxf(d,rt[r+2][c+1]); d=fmaxf(d,rt[r+2][c+2]);
      float nw = fmaxf(cur, d*mt[r][c]);
      nv[k] = nw;
      ch |= (nw != cur);
    }
    __syncthreads();
    if (ch) changed = 1;
    #pragma unroll
    for (int k=0;k<16;k++){
      int p = threadIdx.x + 256*k;
      rt[(p>>6)+1][(p&63)+1] = nv[k];
    }
    __syncthreads();
    if (!changed) break;
  }
  for (int idx=threadIdx.x; idx<64*64; idx+=256){
    int r = idx>>6, c = idx&63;
    rout[(size_t)(ty0+r)*WW + tx0 + c] = rt[r+1][c+1];
  }
}

// ---------------- final combine ----------------
__global__ __launch_bounds__(256) void final_kernel(const float* __restrict__ pred,
                                                    const float* __restrict__ scal,
                                                    const float* __restrict__ reach,
                                                    float* __restrict__ out){
  int im = blockIdx.y;
  size_t idx = (size_t)im*NPIX + blockIdx.x*256 + threadIdx.x;
  float pmn = scal[16+im], pmx = scal[24+im];
  float p = pred[idx];
  if (pmx > 1.f || pmn < 0.f) p = 1.f/(1.f + expf(-p));
  out[idx] = ((p > 0.5f) || (reach[idx] > 0.f)) ? 1.f : 0.f;
}

extern "C" void kernel_launch(void* const* d_in, const int* in_sizes, int n_in,
                              void* d_out, int out_size, void* d_ws, size_t ws_size,
                              hipStream_t stream) {
  (void)in_sizes; (void)n_in; (void)out_size; (void)ws_size;
  const float* pred = (const float*)d_in[0];   // unet_output
  const float* img  = (const float*)d_in[1];   // original_image
  float* out = (float*)d_out;

  float* ws    = (float*)d_ws;
  float* scal  = ws;              // [0:8) img_min [8:16) img_max [16:24) pred_min [24:32) pred_max [32:40) mean
  float* gauss = ws + 64;         // 101 taps
  float* gtap  = ws + 256;        // 4 * 1225 taps
  _Float16* bfrag = (_Float16*)(ws + 8192);   // 4 filters * 70 frags * 512 halfs (32B-aligned)
  const size_t NTOT = (size_t)NIMG*NPIX;
  float* bufA = ws + 81920;       // gauss tmp -> gabor f -> flood pong
  float* bufB = bufA + NTOT;      // enh -> flood ping
  float* bufC = bufB + NTOT;      // seed
  float* bufD = bufC + NTOT;      // grad mask

  init_filters_kernel<<<1,256,0,stream>>>(gauss, gtap);
  init_bfrag_kernel<<<4,256,0,stream>>>(gtap, bfrag);
  minmax_kernel<<<NIMG,1024,0,stream>>>(img,  scal+0,  scal+8);
  minmax_kernel<<<NIMG,1024,0,stream>>>(pred, scal+16, scal+24);

  gauss_h_kernel<<<dim3(WW/256,HH,NIMG),256,0,stream>>>(img, scal, gauss, bufA);
  gauss_v_kernel<<<dim3(WW/256,HH,NIMG),256,0,stream>>>(img, bufA, scal, gauss, bufB);

  const int ksz[4] = {25,35,25,35};   // angles 45,90,135,180 deg
  for (int ff=0; ff<4; ff++){
    gabor_wmma_kernel<<<dim3(WW/64,HH/16,NIMG),32,0,stream>>>(bufB, bfrag + (size_t)ff*70*512, ksz[ff], bufA);
    mean_kernel<<<NIMG,1024,0,stream>>>(bufA, scal+32);
    seed_kernel<<<dim3(NPIX/256,NIMG),256,0,stream>>>(bufA, scal+32, bufC, (ff==0)?1:0);
  }

  sobel_kernel<<<dim3(NPIX/256,NIMG),256,0,stream>>>(bufB, bufD);
  reach0_kernel<<<dim3(NPIX/256,NIMG),256,0,stream>>>(bufC, bufD, bufB);

  for (int p=0; p<24; p++){
    const float* rin = (p & 1) ? bufA : bufB;
    float*       ro  = (p & 1) ? bufB : bufA;
    flood_kernel<<<dim3(WW/64,HH/64,NIMG),256,0,stream>>>(rin, bufD, ro);
  }
  // 24 passes: last (p=23, odd) wrote bufB
  final_kernel<<<dim3(NPIX/256,NIMG),256,0,stream>>>(pred, scal, bufB, out);
}
